// LIPARAttention_31825707664136
// MI455X (gfx1250) — matrix operand order repaired and checked by reading, not compiled, not run on hardware
//
#include <hip/hip_runtime.h>
#include <hip/hip_bf16.h>

typedef __attribute__((ext_vector_type(16))) _Float16 v16h;
typedef __attribute__((ext_vector_type(8)))  _Float16 v8h;
typedef __attribute__((ext_vector_type(4)))  _Float16 v4h;
typedef __attribute__((ext_vector_type(8)))  float    v8f;

#define B_    4
#define N_    4096
#define DIM_  768
#define H_    12
#define DH_   64
#define M_    256
#define SCALE_ 0.125f   // DH^-0.5

// ---------------------------------------------------------------------------
// helpers
// ---------------------------------------------------------------------------
__device__ __forceinline__ v8f vzero8() {
  v8f z;
#pragma unroll
  for (int i = 0; i < 8; i++) z[i] = 0.f;
  return z;
}

__device__ __forceinline__ v8f wmma16(v16h a, v16h b, v8f c) {
  // D(16x16,f32) = A(16x32,f16) * B(32x16,f16) + C
  return __builtin_amdgcn_wmma_f32_16x16x32_f16(false, a, false, b, (short)0, c,
                                                false, false);
}

// A fragment 16x32 f16 (M=row, K=col), row-major source with row stride `stride`.
// lane lo (0..15) = row; half p: k = ((p&8)<<1) + hi*8 + (p&7)  -> two
// contiguous 8-half (16B) chunks at offsets hi*8 and 16+hi*8.
__device__ __forceinline__ v16h load_a_frag(const _Float16* base, int stride,
                                            int k_base, int lo, int hi) {
  const _Float16* p0 = base + (size_t)lo * stride + k_base + hi * 8;
  v8h x0 = *(const v8h*)p0;
  v8h x1 = *(const v8h*)(p0 + 16);
  v16h r;
#pragma unroll
  for (int p = 0; p < 8; p++) { r[p] = x0[p]; r[p + 8] = x1[p]; }
  return r;
}

// B fragment 32x16 f16 (K=row, N=col) read from TRANSPOSED storage BT[n][k]
// with row stride `stride`. lane lo = column n; half p: k = ((p&8)<<1) + hi*8 + (p&7) ->
// two contiguous 8-half (16B) chunks at offsets hi*8 and 16+hi*8.
__device__ __forceinline__ v16h load_bT_frag(const _Float16* baseT, int stride,
                                             int n_base, int k_base, int lo, int hi) {
  const _Float16* p0 = baseT + (size_t)(n_base + lo) * stride + k_base + hi * 8;
  v8h x0 = *(const v8h*)p0;
  v8h x1 = *(const v8h*)(p0 + 16);
  v16h r;
#pragma unroll
  for (int p = 0; p < 8; p++) { r[p] = x0[p]; r[p + 8] = x1[p]; }
  return r;
}

// ---------------------------------------------------------------------------
// Kernel 1: q = x@Wq, kv = x@Wkv  -> f16, head-major (c=b*H+h, n, d) layout.
// 128x128 tile per workgroup, 8 waves as 4(row)x2(col), each wave 32x64.
// ---------------------------------------------------------------------------
__global__ void __launch_bounds__(256)
proj_kernel(const float* __restrict__ x, const float* __restrict__ Wq,
            const float* __restrict__ Wkv, _Float16* __restrict__ qh,
            _Float16* __restrict__ kvh) {
  const float* W  = blockIdx.z ? Wkv : Wq;
  _Float16*   out = blockIdx.z ? kvh : qh;
  const int row0 = blockIdx.x * 128;
  const int col0 = blockIdx.y * 128;
  const int tid = threadIdx.x;
  const int w = tid >> 5, lane = tid & 31, lo = lane & 15, hi = lane >> 4;
  const int wr = w >> 1, wc = w & 1;

  __shared__ __align__(16) _Float16 As[128 * 40];   // [row][k], stride 40
  __shared__ __align__(16) _Float16 BsT[128 * 40];  // [n][k],  stride 40

  v8f acc[2][4];
#pragma unroll
  for (int i = 0; i < 2; i++)
#pragma unroll
    for (int j = 0; j < 4; j++) acc[i][j] = vzero8();

  for (int kb = 0; kb < 24; kb++) {
    // cooperative A tile: x[row0+row][kb*32 + c4..c4+3]  (f32 -> f16)
    {
      int rbase = tid >> 3;
      int c4 = (tid & 7) * 4;
#pragma unroll
      for (int rr = 0; rr < 4; rr++) {
        int row = rbase + rr * 32;
        float4 v = *(const float4*)(x + (size_t)(row0 + row) * DIM_ + kb * 32 + c4);
        _Float16* dst = &As[row * 40 + c4];
        dst[0] = (_Float16)v.x; dst[1] = (_Float16)v.y;
        dst[2] = (_Float16)v.z; dst[3] = (_Float16)v.w;
      }
      if (kb + 1 < 24)
        __builtin_prefetch(x + (size_t)(row0 + rbase) * DIM_ + (kb + 1) * 32 + c4, 0, 0);
      // cooperative B tile, stored transposed: BsT[n][kk] = W[kb*32+kk][col0+n]
      int kkb = tid >> 5;
      int n4 = (tid & 31) * 4;
#pragma unroll
      for (int rr = 0; rr < 4; rr++) {
        int kk = kkb + rr * 8;
        float4 v = *(const float4*)(W + (size_t)(kb * 32 + kk) * DIM_ + col0 + n4);
        BsT[(n4 + 0) * 40 + kk] = (_Float16)v.x;
        BsT[(n4 + 1) * 40 + kk] = (_Float16)v.y;
        BsT[(n4 + 2) * 40 + kk] = (_Float16)v.z;
        BsT[(n4 + 3) * 40 + kk] = (_Float16)v.w;
      }
    }
    __syncthreads();

    v16h a0 = load_a_frag(&As[(wr * 32) * 40], 40, 0, lo, hi);
    v16h a1 = load_a_frag(&As[(wr * 32 + 16) * 40], 40, 0, lo, hi);
#pragma unroll
    for (int j = 0; j < 4; j++) {
      v16h bf = load_bT_frag(BsT, 40, wc * 64 + j * 16, 0, lo, hi);
      acc[0][j] = wmma16(a0, bf, acc[0][j]);
      acc[1][j] = wmma16(a1, bf, acc[1][j]);
    }
    __syncthreads();
  }

  // epilogue: scatter into head-major f16
#pragma unroll
  for (int i = 0; i < 2; i++)
#pragma unroll
    for (int j = 0; j < 4; j++)
#pragma unroll
      for (int r = 0; r < 8; r++) {
        int t  = row0 + wr * 32 + i * 16 + r + hi * 8;   // global token row
        int jj = col0 + wc * 64 + j * 16 + lo;           // global out col
        int b = t >> 12, n = t & (N_ - 1);
        int h = jj >> 6, d = jj & 63;
        out[(((size_t)b * H_ + h) * N_ + n) * DH_ + d] = (_Float16)acc[i][j][r];
      }
}

// ---------------------------------------------------------------------------
// Kernel 2: in-place rotary on (d, d+32) pairs; q additionally * SCALE.
// one thread per (c, n, d<32) pair.
// ---------------------------------------------------------------------------
__global__ void __launch_bounds__(256)
rotary_kernel(_Float16* __restrict__ qh, _Float16* __restrict__ kvh) {
  const long idx = (long)blockIdx.x * 256 + threadIdx.x;  // < 48*4096*32
  const int d = (int)(idx & 31);
  const long cn = idx >> 5;            // c*N + n
  const int n = (int)(cn & (N_ - 1));
  // inv_freq = 10000^(-2d/64) = exp(-d * ln(10000)/32)
  const float inv = __expf(-(float)d * 0.28782313662425576f);
  float sn, cs;
  __sincosf((float)n * inv, &sn, &cs);
  const size_t base = (size_t)cn * DH_ + d;
  float q1 = (float)qh[base],  q2 = (float)qh[base + 32];
  float k1 = (float)kvh[base], k2 = (float)kvh[base + 32];
  qh[base]       = (_Float16)((q1 * cs - q2 * sn) * SCALE_);
  qh[base + 32]  = (_Float16)((q2 * cs + q1 * sn) * SCALE_);
  kvh[base]      = (_Float16)(k1 * cs - k2 * sn);
  kvh[base + 32] = (_Float16)(k2 * cs + k1 * sn);
}

// ---------------------------------------------------------------------------
// Kernel 3: segment attention, flash-style online softmax.
// grid (c=48, s=16); block 256 = 8 waves; wave owns 32 query rows.
// s==0: keys = rows [0,256); s>=1: keys = rows [(s-1)*256, (s+1)*256).
// ---------------------------------------------------------------------------
__global__ void __launch_bounds__(256)
attn_kernel(const _Float16* __restrict__ qh, const _Float16* __restrict__ kvh,
            _Float16* __restrict__ att) {
  const int c = blockIdx.x;
  const int s = blockIdx.y;
  const int tid = threadIdx.x;
  const int w = tid >> 5;
  const int lane = tid & 31;
  const int lo = lane & 15, hi = lane >> 4;

  const int n0 = s * M_;
  const int kstart = (s == 0) ? 0 : (s - 1) * M_;
  const int nchunks = (s == 0) ? 8 : 16;   // 32-key chunks

  __shared__ __align__(16) _Float16 kv_lds[32 * 72];   // [key][d],  stride 72
  __shared__ __align__(16) _Float16 kv_ldsT[64 * 40];  // [d][key],  stride 40
  __shared__ __align__(16) _Float16 p_lds[8 * 32 * 32];// per wave [row][key]

  // persistent Q A-fragments (2 row tiles x 2 K-steps over DH=64)
  const size_t qbase = ((size_t)c * N_ + n0 + w * 32) * DH_;
  v16h qa[2][2];
#pragma unroll
  for (int i = 0; i < 2; i++)
#pragma unroll
    for (int kk = 0; kk < 2; kk++)
      qa[i][kk] = load_a_frag(qh + qbase + (size_t)i * 16 * DH_, DH_, kk * 32, lo, hi);

  v8f o[2][4];
  float m_st[2][8], l_st[2][8];
#pragma unroll
  for (int i = 0; i < 2; i++) {
#pragma unroll
    for (int j = 0; j < 4; j++) o[i][j] = vzero8();
#pragma unroll
    for (int r = 0; r < 8; r++) { m_st[i][r] = -1e30f; l_st[i][r] = 0.f; }
  }

  for (int ch = 0; ch < nchunks; ch++) {
    // cooperative kv chunk load: 32 keys x 64 d, both layouts
    {
      int key = tid >> 3;
      int d0 = (tid & 7) * 8;
      v8h v = *(const v8h*)(kvh + ((size_t)c * N_ + kstart + ch * 32 + key) * DH_ + d0);
      *(v8h*)&kv_lds[key * 72 + d0] = v;
#pragma unroll
      for (int u = 0; u < 8; u++) kv_ldsT[(d0 + u) * 40 + key] = v[u];
    }
    __syncthreads();

    // S = Q Kc^T  (2 row tiles x 2 key tiles, K-dim = 64 via 2 WMMA)
    v8f sf[2][2];
#pragma unroll
    for (int i = 0; i < 2; i++)
#pragma unroll
      for (int j = 0; j < 2; j++) sf[i][j] = vzero8();
#pragma unroll
    for (int kk = 0; kk < 2; kk++)
#pragma unroll
      for (int j = 0; j < 2; j++) {
        v16h bf = load_bT_frag(kv_lds, 72, j * 16, kk * 32, lo, hi);
#pragma unroll
        for (int i = 0; i < 2; i++) sf[i][j] = wmma16(qa[i][kk], bf, sf[i][j]);
      }

    // online softmax; C-layout row = r + hi*8 matches per-lane stats + width-16 shuffles
#pragma unroll
    for (int i = 0; i < 2; i++) {
#pragma unroll
      for (int r = 0; r < 8; r++) {
        float mx = fmaxf(sf[i][0][r], sf[i][1][r]);
#pragma unroll
        for (int off = 1; off < 16; off <<= 1) mx = fmaxf(mx, __shfl_xor(mx, off, 16));
        float mn = fmaxf(m_st[i][r], mx);
        float sc = __expf(m_st[i][r] - mn);
        m_st[i][r] = mn;
        float rs = 0.f;
#pragma unroll
        for (int j = 0; j < 2; j++) {
          float p = __expf(sf[i][j][r] - mn);
          sf[i][j][r] = p;
          rs += p;
        }
#pragma unroll
        for (int off = 1; off < 16; off <<= 1) rs += __shfl_xor(rs, off, 16);
        l_st[i][r] = l_st[i][r] * sc + rs;
#pragma unroll
        for (int j = 0; j < 4; j++) o[i][j][r] *= sc;
      }
      // P -> per-wave LDS (C layout -> row-major) for A-fragment reload
#pragma unroll
      for (int j = 0; j < 2; j++)
#pragma unroll
        for (int r = 0; r < 8; r++)
          p_lds[w * 1024 + (i * 16 + r + hi * 8) * 32 + j * 16 + lo] =
              (_Float16)sf[i][j][r];
    }

    // O += P @ Vc   (K = 32 keys, one WMMA per 16x16 tile, 4 d-tiles)
#pragma unroll
    for (int i = 0; i < 2; i++) {
      v16h pa = load_a_frag(&p_lds[w * 1024 + i * 16 * 32], 32, 0, lo, hi);
#pragma unroll
      for (int j = 0; j < 4; j++) {
        v16h bf = load_bT_frag(kv_ldsT, 40, j * 16, 0, lo, hi);
        o[i][j] = wmma16(pa, bf, o[i][j]);
      }
    }
    __syncthreads();
  }

  // normalize and store head-major f16
#pragma unroll
  for (int i = 0; i < 2; i++)
#pragma unroll
    for (int r = 0; r < 8; r++) {
      float inv = 1.f / l_st[i][r];
      size_t row = (size_t)c * N_ + n0 + w * 32 + i * 16 + r + hi * 8;
#pragma unroll
      for (int j = 0; j < 4; j++)
        att[row * DH_ + j * 16 + lo] = (_Float16)(o[i][j][r] * inv);
    }
}

// ---------------------------------------------------------------------------
// Kernel 4: out = att(merged heads) @ {Wo|Wo0} + {bo|bo0}, f32 output.
// Per-batch tokens <256 use Wo0/bo0; 128-row blocks never straddle the split.
// ---------------------------------------------------------------------------
__global__ void __launch_bounds__(256)
oproj_kernel(const _Float16* __restrict__ att, const float* __restrict__ Wo,
             const float* __restrict__ bo, const float* __restrict__ Wo0,
             const float* __restrict__ bo0, float* __restrict__ out) {
  const int row0 = blockIdx.x * 128;
  const int col0 = blockIdx.y * 128;
  const bool seg0 = (row0 & (N_ - 1)) < M_;
  const float* W    = seg0 ? Wo0 : Wo;
  const float* bias = seg0 ? bo0 : bo;
  const int tid = threadIdx.x;
  const int w = tid >> 5, lane = tid & 31, lo = lane & 15, hi = lane >> 4;
  const int wr = w >> 1, wc = w & 1;

  __shared__ __align__(16) _Float16 As[128 * 40];
  __shared__ __align__(16) _Float16 BsT[128 * 40];

  v8f acc[2][4];
#pragma unroll
  for (int i = 0; i < 2; i++)
#pragma unroll
    for (int j = 0; j < 4; j++) acc[i][j] = vzero8();

  for (int kb = 0; kb < 24; kb++) {
    const int h = kb >> 1;
    const int dbase = (kb & 1) * 32;  // 32-wide K block sits inside one head
    {
      int rbase = tid >> 3;
      int c4 = (tid & 7) * 4;
#pragma unroll
      for (int rr = 0; rr < 4; rr++) {
        int row = rbase + rr * 32;
        int t = row0 + row;
        int b = t >> 12, n = t & (N_ - 1);
        v4h v = *(const v4h*)(att + (((size_t)b * H_ + h) * N_ + n) * DH_ + dbase + c4);
        _Float16* dst = &As[row * 40 + c4];
        dst[0] = v[0]; dst[1] = v[1]; dst[2] = v[2]; dst[3] = v[3];
      }
      int kkb = tid >> 5;
      int n4 = (tid & 31) * 4;
#pragma unroll
      for (int rr = 0; rr < 4; rr++) {
        int kk = kkb + rr * 8;
        float4 v = *(const float4*)(W + (size_t)(kb * 32 + kk) * DIM_ + col0 + n4);
        BsT[(n4 + 0) * 40 + kk] = (_Float16)v.x;
        BsT[(n4 + 1) * 40 + kk] = (_Float16)v.y;
        BsT[(n4 + 2) * 40 + kk] = (_Float16)v.z;
        BsT[(n4 + 3) * 40 + kk] = (_Float16)v.w;
      }
      if (kb + 1 < 24)
        __builtin_prefetch(W + (size_t)((kb + 1) * 32 + kkb) * DIM_ + col0 + n4, 0, 0);
    }
    __syncthreads();

    v16h a0 = load_a_frag(&As[(wr * 32) * 40], 40, 0, lo, hi);
    v16h a1 = load_a_frag(&As[(wr * 32 + 16) * 40], 40, 0, lo, hi);
#pragma unroll
    for (int j = 0; j < 4; j++) {
      v16h bf = load_bT_frag(BsT, 40, wc * 64 + j * 16, 0, lo, hi);
      acc[0][j] = wmma16(a0, bf, acc[0][j]);
      acc[1][j] = wmma16(a1, bf, acc[1][j]);
    }
    __syncthreads();
  }

#pragma unroll
  for (int i = 0; i < 2; i++)
#pragma unroll
    for (int j = 0; j < 4; j++)
#pragma unroll
      for (int r = 0; r < 8; r++) {
        int t  = row0 + wr * 32 + i * 16 + r + hi * 8;
        int jj = col0 + wc * 64 + j * 16 + lo;
        out[(size_t)t * DIM_ + jj] = acc[i][j][r] + bias[jj];
      }
}

// ---------------------------------------------------------------------------
extern "C" void kernel_launch(void* const* d_in, const int* in_sizes, int n_in,
                              void* d_out, int out_size, void* d_ws, size_t ws_size,
                              hipStream_t stream) {
  const float* x   = (const float*)d_in[0];
  const float* Wq  = (const float*)d_in[1];
  const float* Wkv = (const float*)d_in[2];
  const float* Wo  = (const float*)d_in[3];
  const float* bo  = (const float*)d_in[4];
  const float* Wo0 = (const float*)d_in[5];
  const float* bo0 = (const float*)d_in[6];
  float* out = (float*)d_out;

  const size_t HEAD_ELEMS = (size_t)B_ * H_ * N_ * DH_;  // 12,582,912
  _Float16* qh  = (_Float16*)d_ws;
  _Float16* kvh = qh  + HEAD_ELEMS;
  _Float16* att = kvh + HEAD_ELEMS;

  // 1) projections -> head-major f16
  proj_kernel<<<dim3(128, 6, 2), 256, 0, stream>>>(x, Wq, Wkv, qh, kvh);
  // 2) rotary (+ q scale), in place
  rotary_kernel<<<(B_ * H_ * N_ * (DH_ / 2)) / 256, 256, 0, stream>>>(qh, kvh);
  // 3) segment attention
  attn_kernel<<<dim3(48, 16), 256, 0, stream>>>(qh, kvh, att);
  // 4) output projection + bias
  oproj_kernel<<<dim3(128, 6), 256, 0, stream>>>(att, Wo, bo, Wo0, bo0, out);
}
